// JambaMoE_10445360464008
// MI455X (gfx1250) — matrix operand order repaired and checked by reading, compiled, not run on hardware
//
#include <hip/hip_runtime.h>
#include <cstdint>

// ---------------- problem constants (fixed by reference) ----------------
#define T_TOK 8192
#define H_DIM 2048
#define I_DIM 4096
#define E_NUM 8
#define LDA   40                 // padded LDS row stride (bf16): 80B rows -> conflict-free b128 frags
#define TILE_ELEMS (128 * LDA)

typedef __bf16 bf16;
typedef __attribute__((ext_vector_type(16))) __bf16 v16bf;
typedef __attribute__((ext_vector_type(8)))  __bf16 v8bf;
typedef __attribute__((ext_vector_type(4)))  __bf16 v4bf;
typedef __attribute__((ext_vector_type(8)))  float  v8f;
typedef __attribute__((ext_vector_type(4)))  float  v4f;

__device__ __forceinline__ v16bf cat8(v8bf lo, v8bf hi) {
  return __builtin_shufflevector(lo, hi, 0,1,2,3,4,5,6,7,8,9,10,11,12,13,14,15);
}

// per-lane async copy: 16B global -> LDS, tracked by ASYNCcnt (CDNA5 path)
__device__ __forceinline__ void async_copy_b128(const void* gptr, void* lptr) {
  unsigned lds32 = (unsigned)(uintptr_t)lptr;               // LDS addr = generic[31:0]
  asm volatile("global_load_async_to_lds_b128 %0, %1, off"
               :: "v"(lds32), "v"((unsigned long long)(uintptr_t)gptr)
               : "memory");
}
__device__ __forceinline__ void async_wait0() {
  asm volatile("s_wait_asynccnt 0x0" ::: "memory");
}

// ---------------- 0) one-shot fp32 -> bf16 conversion (activations) ----------------
__global__ __launch_bounds__(256) void cvt_bf16(const float* __restrict__ src,
                                                bf16* __restrict__ dst) {
  size_t i = ((size_t)blockIdx.x * 256 + threadIdx.x) * 4;
  v4f v = *(const v4f*)(src + i);
  v4bf b = { (bf16)v[0], (bf16)v[1], (bf16)v[2], (bf16)v[3] };
  *(v4bf*)(dst + i) = b;
}

// ---------------- 1) router: logits -> softmax -> top-2 ----------------
__global__ __launch_bounds__(256) void moe_router(
    const float* __restrict__ x, const float* __restrict__ rw,
    int* __restrict__ topi, float* __restrict__ topw) {
  const int wid = threadIdx.x >> 5, lane = threadIdx.x & 31;
  const int t = blockIdx.x * 8 + wid;
  float acc[E_NUM];
#pragma unroll
  for (int e = 0; e < E_NUM; ++e) acc[e] = 0.f;
  const float* xr = x + (size_t)t * H_DIM;
  for (int h = lane; h < H_DIM; h += 32) {
    float xv = xr[h];
#pragma unroll
    for (int e = 0; e < E_NUM; ++e) acc[e] += xv * rw[e * H_DIM + h];
  }
#pragma unroll
  for (int e = 0; e < E_NUM; ++e) {
#pragma unroll
    for (int off = 16; off > 0; off >>= 1) acc[e] += __shfl_xor(acc[e], off, 32);
  }
  if (lane == 0) {
    float m = acc[0];
#pragma unroll
    for (int e = 1; e < E_NUM; ++e) m = fmaxf(m, acc[e]);
    float p[E_NUM]; float s = 0.f;
#pragma unroll
    for (int e = 0; e < E_NUM; ++e) { p[e] = __expf(acc[e] - m); s += p[e]; }
    float inv = 1.f / s;
#pragma unroll
    for (int e = 0; e < E_NUM; ++e) p[e] *= inv;
    int i0 = 0; float p0 = p[0];
#pragma unroll
    for (int e = 1; e < E_NUM; ++e) if (p[e] > p0) { p0 = p[e]; i0 = e; }
    int i1 = -1; float p1 = -1.f;
#pragma unroll
    for (int e = 0; e < E_NUM; ++e) if (e != i0 && p[e] > p1) { p1 = p[e]; i1 = e; }
    topi[2 * t]     = i0;  topi[2 * t + 1] = i1;
    topw[2 * t]     = p0;  topw[2 * t + 1] = p1;
  }
}

// ---------------- 2) deterministic per-expert compaction ----------------
__global__ __launch_bounds__(256) void moe_compact(
    const int* __restrict__ topi, const float* __restrict__ topw,
    int* __restrict__ tlist, float* __restrict__ tw, int* __restrict__ cnt) {
  const int w = threadIdx.x >> 5, lane = threadIdx.x & 31;
  int offset = 0;
  for (int base = 0; base < T_TOK; base += 32) {
    int t = base + lane;
    int i0 = topi[2 * t], i1 = topi[2 * t + 1];
    bool f0 = (i0 == w), f1 = (i1 == w);
    bool f = f0 || f1;
    float p = f0 ? topw[2 * t] : topw[2 * t + 1];
    unsigned mask = (unsigned)__ballot(f);
    int slot = offset + __popc(mask & ((1u << lane) - 1u));
    if (f) {
      tlist[(size_t)w * T_TOK + slot] = t;
      tw[(size_t)w * T_TOK + slot] = p;
    }
    offset += __popc(mask);
  }
  if (lane == 0) cnt[w] = offset;
}

// ---------------- 3) gathered gate/up GEMM + fused SwiGLU ----------------
// Double-buffered LDS: async A-tile DMA + B global loads for tile k+1 issued
// before tile k's WMMAs; convert+ds_store after; one barrier per k-step.
__global__ __launch_bounds__(256) void moe_gateup(
    const bf16* __restrict__ xb, const float* __restrict__ wsE,
    const int* __restrict__ tl, const int* __restrict__ cntE,
    bf16* __restrict__ hb) {
  const int n = *cntE;
  if ((int)blockIdx.x * 128 >= n) return;

  __shared__ bf16 As[2][TILE_ELEMS], Bg[2][TILE_ELEMS], Bu[2][TILE_ELEMS];
  const int tid = threadIdx.x, lane = tid & 31, wid = tid >> 5;
  const int wM = wid >> 2, wN = wid & 3;
  const int l15 = lane & 15, hi = lane >> 4;
  const int rowTile = blockIdx.x * 128;
  const int colBase = blockIdx.y * 128;
  const size_t upOff = (size_t)I_DIM * H_DIM;

  // ---- k-invariant staging descriptors (hoisted: no token reload in the loop) ----
  const bf16* aSrc[2]; int aDst[2];
#pragma unroll
  for (int it = 0; it < 2; ++it) {
    int q = tid + it * 256;                  // 512 x 16B chunks
    int row = q >> 2, c8 = (q & 3) * 8;
    int slot = rowTile + row;
    int tok = tl[slot < n ? slot : (n - 1)];
    aSrc[it] = xb + (size_t)tok * H_DIM + c8;
    aDst[it] = row * LDA + c8;
  }
  const float* bSrc[4]; int bDst[4];
#pragma unroll
  for (int it = 0; it < 4; ++it) {
    int q = tid + it * 256;
    int row = q >> 3, c4 = (q & 7) * 4;
    bSrc[it] = wsE + (size_t)(colBase + row) * H_DIM + c4;
    bDst[it] = row * LDA + c4;
  }

  v8f cg[4][2], cu[4][2];
#pragma unroll
  for (int mi = 0; mi < 4; ++mi)
#pragma unroll
    for (int ni = 0; ni < 2; ++ni) { cg[mi][ni] = (v8f)(0.f); cu[mi][ni] = (v8f)(0.f); }

  v4f vg[4], vu[4];
  auto stageA = [&](int b, int kk) {
#pragma unroll
    for (int it = 0; it < 2; ++it)
      async_copy_b128(aSrc[it] + kk, &As[b][aDst[it]]);
  };
  auto loadB = [&](int kk) {
#pragma unroll
    for (int it = 0; it < 4; ++it) {
      vg[it] = *(const v4f*)(bSrc[it] + kk);
      vu[it] = *(const v4f*)(bSrc[it] + upOff + kk);
      if (kk + 64 < H_DIM) {                 // two tiles ahead (global_prefetch_b8)
        __builtin_prefetch(bSrc[it] + kk + 64, 0, 3);
        __builtin_prefetch(bSrc[it] + upOff + kk + 64, 0, 3);
      }
    }
  };
  auto storeB = [&](int b) {
#pragma unroll
    for (int it = 0; it < 4; ++it) {
      v4bf g = { (bf16)vg[it][0], (bf16)vg[it][1], (bf16)vg[it][2], (bf16)vg[it][3] };
      v4bf u = { (bf16)vu[it][0], (bf16)vu[it][1], (bf16)vu[it][2], (bf16)vu[it][3] };
      *(v4bf*)(&Bg[b][bDst[it]]) = g;
      *(v4bf*)(&Bu[b][bDst[it]]) = u;
    }
  };

  // prologue: fill buffer 0
  stageA(0, 0);
  loadB(0);
  storeB(0);
  async_wait0();
  __syncthreads();

  int buf = 0;
  for (int kk = 32; kk <= H_DIM; kk += 32, buf ^= 1) {
    const bool more = (kk < H_DIM);
    if (more) { stageA(buf ^ 1, kk); loadB(kk); }   // next tile in flight

    const bf16* Ab  = As[buf];
    const bf16* Bgb = Bg[buf];
    const bf16* Bub = Bu[buf];
    v16bf a[4], bgf[2], buf2[2];
#pragma unroll
    for (int mi = 0; mi < 4; ++mi) {   // A frag: K 0-7 / 8-15 by lane half, +16 in upper VGPRs
      int r = wM * 64 + mi * 16 + l15;
      int kb = hi * 8;
      a[mi] = cat8(*(const v8bf*)(Ab + r * LDA + kb),
                   *(const v8bf*)(Ab + r * LDA + kb + 16));
    }
#pragma unroll
    for (int ni = 0; ni < 2; ++ni) {   // B frag: K 0-7 / 8-15 by lane half, +16 in upper VGPRs
      int r = wN * 32 + ni * 16 + l15;
      int kb = hi * 8;
      bgf[ni]  = cat8(*(const v8bf*)(Bgb + r * LDA + kb), *(const v8bf*)(Bgb + r * LDA + kb + 16));
      buf2[ni] = cat8(*(const v8bf*)(Bub + r * LDA + kb), *(const v8bf*)(Bub + r * LDA + kb + 16));
    }
#pragma unroll
    for (int mi = 0; mi < 4; ++mi)
#pragma unroll
      for (int ni = 0; ni < 2; ++ni) {
        cg[mi][ni] = __builtin_amdgcn_wmma_f32_16x16x32_bf16(false, a[mi], false, bgf[ni],
                                                             (short)0, cg[mi][ni], false, false);
        cu[mi][ni] = __builtin_amdgcn_wmma_f32_16x16x32_bf16(false, a[mi], false, buf2[ni],
                                                             (short)0, cu[mi][ni], false, false);
      }
    if (more) { storeB(buf ^ 1); async_wait0(); }
    __syncthreads();
  }

  // fused SwiGLU epilogue -> bf16 h (fast sigmoid via v_rcp_f32)
#pragma unroll
  for (int mi = 0; mi < 4; ++mi)
#pragma unroll
    for (int r = 0; r < 8; ++r) {
      int slot = rowTile + wM * 64 + mi * 16 + hi * 8 + r;
      if (slot < n) {
#pragma unroll
        for (int ni = 0; ni < 2; ++ni) {
          int col = colBase + wN * 32 + ni * 16 + l15;
          float g = cg[mi][ni][r], u = cu[mi][ni][r];
          float h = g * __builtin_amdgcn_rcpf(1.f + __expf(-g)) * u;
          hb[(size_t)slot * I_DIM + col] = (bf16)h;
        }
      }
    }
}

// ---------------- 4) down projection + weighted scatter ----------------
__global__ __launch_bounds__(256) void moe_down(
    const bf16* __restrict__ hb, const float* __restrict__ w2E,
    const int* __restrict__ tl, const float* __restrict__ twE,
    const int* __restrict__ cntE, float* __restrict__ out) {
  const int n = *cntE;
  if ((int)blockIdx.x * 128 >= n) return;

  __shared__ bf16 As[2][TILE_ELEMS], Bs[2][TILE_ELEMS];
  const int tid = threadIdx.x, lane = tid & 31, wid = tid >> 5;
  const int wM = wid >> 2, wN = wid & 3;
  const int l15 = lane & 15, hi = lane >> 4;
  const int rowTile = blockIdx.x * 128;
  const int colBase = blockIdx.y * 128;

  const bf16* aSrc[2]; int aDst[2];
#pragma unroll
  for (int it = 0; it < 2; ++it) {
    int q = tid + it * 256;
    int row = q >> 2, c8 = (q & 3) * 8;
    aSrc[it] = hb + (size_t)(rowTile + row) * I_DIM + c8;
    aDst[it] = row * LDA + c8;
  }
  const float* bSrc[4]; int bDst[4];
#pragma unroll
  for (int it = 0; it < 4; ++it) {
    int q = tid + it * 256;
    int row = q >> 3, c4 = (q & 7) * 4;
    bSrc[it] = w2E + (size_t)(colBase + row) * I_DIM + c4;
    bDst[it] = row * LDA + c4;
  }

  v8f c[4][2];
#pragma unroll
  for (int mi = 0; mi < 4; ++mi)
#pragma unroll
    for (int ni = 0; ni < 2; ++ni) c[mi][ni] = (v8f)(0.f);

  v4f vb[4];
  auto stageA = [&](int b, int kk) {
#pragma unroll
    for (int it = 0; it < 2; ++it)
      async_copy_b128(aSrc[it] + kk, &As[b][aDst[it]]);
  };
  auto loadB = [&](int kk) {
#pragma unroll
    for (int it = 0; it < 4; ++it) {
      vb[it] = *(const v4f*)(bSrc[it] + kk);
      if (kk + 64 < I_DIM) __builtin_prefetch(bSrc[it] + kk + 64, 0, 3);
    }
  };
  auto storeB = [&](int b) {
#pragma unroll
    for (int it = 0; it < 4; ++it) {
      v4bf v = { (bf16)vb[it][0], (bf16)vb[it][1], (bf16)vb[it][2], (bf16)vb[it][3] };
      *(v4bf*)(&Bs[b][bDst[it]]) = v;
    }
  };

  stageA(0, 0);
  loadB(0);
  storeB(0);
  async_wait0();
  __syncthreads();

  int buf = 0;
  for (int kk = 32; kk <= I_DIM; kk += 32, buf ^= 1) {
    const bool more = (kk < I_DIM);
    if (more) { stageA(buf ^ 1, kk); loadB(kk); }

    const bf16* Ab = As[buf];
    const bf16* Bb = Bs[buf];
    v16bf a[4], bf_[2];
#pragma unroll
    for (int mi = 0; mi < 4; ++mi) {
      int r = wM * 64 + mi * 16 + l15;
      int kb = hi * 8;
      a[mi] = cat8(*(const v8bf*)(Ab + r * LDA + kb),
                   *(const v8bf*)(Ab + r * LDA + kb + 16));
    }
#pragma unroll
    for (int ni = 0; ni < 2; ++ni) {
      int r = wN * 32 + ni * 16 + l15;
      int kb = hi * 8;
      bf_[ni] = cat8(*(const v8bf*)(Bb + r * LDA + kb), *(const v8bf*)(Bb + r * LDA + kb + 16));
    }
#pragma unroll
    for (int mi = 0; mi < 4; ++mi)
#pragma unroll
      for (int ni = 0; ni < 2; ++ni)
        c[mi][ni] = __builtin_amdgcn_wmma_f32_16x16x32_bf16(false, a[mi], false, bf_[ni],
                                                            (short)0, c[mi][ni], false, false);
    if (more) { storeB(buf ^ 1); async_wait0(); }
    __syncthreads();
  }

#pragma unroll
  for (int mi = 0; mi < 4; ++mi)
#pragma unroll
    for (int r = 0; r < 8; ++r) {
      int slot = rowTile + wM * 64 + mi * 16 + hi * 8 + r;
      if (slot < n) {
        int tok = tl[slot];
        float wgt = twE[slot];
#pragma unroll
        for (int ni = 0; ni < 2; ++ni) {
          int col = colBase + wN * 32 + ni * 16 + l15;
          size_t idx = (size_t)tok * H_DIM + col;
          out[idx] += wgt * c[mi][ni][r];
        }
      }
    }
}

// ---------------- host orchestration ----------------
extern "C" void kernel_launch(void* const* d_in, const int* in_sizes, int n_in,
                              void* d_out, int out_size, void* d_ws, size_t ws_size,
                              hipStream_t stream) {
  (void)in_sizes; (void)n_in; (void)out_size; (void)ws_size;
  const float* x  = (const float*)d_in[0];
  const float* rw = (const float*)d_in[1];
  const float* ws = (const float*)d_in[2];
  const float* w2 = (const float*)d_in[3];
  float* out = (float*)d_out;

  char* wsp = (char*)d_ws;
  int*   topi  = (int*)(wsp + 0);                        //  64 KB
  float* topw  = (float*)(wsp + 65536);                  //  64 KB
  int*   tlist = (int*)(wsp + 131072);                   // 256 KB
  float* tw    = (float*)(wsp + 393216);                 // 256 KB
  int*   cnt   = (int*)(wsp + 655360);                   //  32 B
  bf16*  xb    = (bf16*)(wsp + (1ull << 20));            //  32 MB
  bf16*  hb    = (bf16*)(wsp + (33ull << 20));           //  64 MB

  cvt_bf16<<<dim3((T_TOK * H_DIM) / (256 * 4)), 256, 0, stream>>>(x, xb);
  moe_router<<<dim3(T_TOK / 8), 256, 0, stream>>>(x, rw, topi, topw);
  moe_compact<<<dim3(1), 256, 0, stream>>>(topi, topw, tlist, tw, cnt);
  hipMemsetAsync(d_out, 0, (size_t)T_TOK * H_DIM * sizeof(float), stream);

  for (int e = 0; e < E_NUM; ++e) {
    moe_gateup<<<dim3(T_TOK / 128, I_DIM / 128), 256, 0, stream>>>(
        xb, ws + (size_t)e * 2 * I_DIM * H_DIM, tlist + (size_t)e * T_TOK, cnt + e, hb);
    moe_down<<<dim3(T_TOK / 128, H_DIM / 128), 256, 0, stream>>>(
        hb, w2 + (size_t)e * H_DIM * I_DIM, tlist + (size_t)e * T_TOK,
        tw + (size_t)e * T_TOK, cnt + e, out);
  }
}
